// MMDLoss_34385508172276
// MI455X (gfx1250) — matrix operand order repaired and checked by reading, not compiled, not run on hardware
//
#include <hip/hip_runtime.h>
#include <hip/hip_bf16.h>

// ---------------------------------------------------------------------------
// MMD loss, MI455X (gfx1250, wave32).
//   * Gram matrix via V_WMMA_F32_16X16X32_BF16, bf16 hi/lo split (3 WMMA per
//     32-K chunk => near-fp32 accuracy at bf16 matrix-pipe rate).
//   * Bandwidth analytically: sum(L2) = 2n*sum|x|^2 - 2*|sum x|^2.
//   * Fast path: fp32->bf16 hi/lo planes precomputed in workspace, Gram tiles
//     staged with GLOBAL_LOAD_ASYNC_TO_LDS_B128 (double buffered, ASYNCcnt).
//   * LDS rows padded to 80B: 16B-aligned ds_load_b128 operand fetch,
//     conflict-free (20*m mod 64 distinct).
// ---------------------------------------------------------------------------

typedef __attribute__((ext_vector_type(16))) __bf16        v16bf;
typedef __attribute__((ext_vector_type(8)))  float         v8f;
typedef __attribute__((ext_vector_type(8)))  unsigned int  v8u;
typedef __attribute__((ext_vector_type(4)))  unsigned int  v4u;
typedef int vi4 __attribute__((vector_size(16)));   // matches async-LDS builtin param

#define N_ROWS   4096
#define DIMS     256
#define TOT      8192          // 2N
#define TILE     64            // 64x64 output tile per workgroup (4 waves)
#define KC       32            // K chunk per WMMA round
#define LSTRIDE  40            // ushorts per LDS row (80B): 16B aligned, no conflicts
#define SURF     (TILE * LSTRIDE)   // ushorts per surface (Ahi/Alo/Bhi/Blo)
#define CHUNK    (4 * SURF)         // ushorts per staged chunk

// workspace layout
#define WS_SUMSQ   0
#define WS_RES     1
#define WS_COEF    2           // 5 floats: -1/(bw*2^k)
#define WS_COLSUM  8           // 256 floats
#define WS_SQ      264         // 8192 floats
#define WS_FLOATS  (WS_SQ + TOT)
#define WS_HILO_BYTES ((size_t)2 * TOT * DIMS * 2)   // HI + LO bf16 planes

__device__ __forceinline__ const float* row_ptr(const float* __restrict__ src,
                                                const float* __restrict__ tgt,
                                                int i) {
  return (i < N_ROWS) ? (src + (size_t)i * DIMS)
                      : (tgt + (size_t)(i - N_ROWS) * DIMS);
}

__device__ __forceinline__ unsigned short bf16_rne(float x) {
  unsigned int u = __float_as_uint(x);
  unsigned int r = u + 0x7FFFu + ((u >> 16) & 1u);
  return (unsigned short)(r >> 16);
}
__device__ __forceinline__ float bf16_up(unsigned short h) {
  return __uint_as_float(((unsigned int)h) << 16);
}

// ---- CDNA5 async global->LDS copy (16B), with compile-safe fallback --------
__device__ __forceinline__ void async_copy16(const void* g, void* l) {
#if defined(__has_builtin) && __has_builtin(__builtin_amdgcn_global_load_async_to_lds_b128)
  __builtin_amdgcn_global_load_async_to_lds_b128(
      (__attribute__((address_space(1))) vi4*)(unsigned long long)g,
      (__attribute__((address_space(3))) vi4*)(unsigned int)(unsigned long long)l,
      0, 0);
#else
  *(v4u*)l = *(const v4u*)g;
#endif
}
__device__ __forceinline__ void wait_async_le8() {
#if defined(__has_builtin) && __has_builtin(__builtin_amdgcn_s_wait_asynccnt)
  __builtin_amdgcn_s_wait_asynccnt(8);
#else
  asm volatile("s_wait_asynccnt 0x8" ::: "memory");
#endif
}
__device__ __forceinline__ void wait_async_le0() {
#if defined(__has_builtin) && __has_builtin(__builtin_amdgcn_s_wait_asynccnt)
  __builtin_amdgcn_s_wait_asynccnt(0);
#else
  asm volatile("s_wait_asynccnt 0x0" ::: "memory");
#endif
}

// ------------------------- kernel 0: zero accumulators ----------------------
__global__ void mmd_zero_kernel(float* __restrict__ ws, float* __restrict__ out) {
  int i = blockIdx.x * blockDim.x + threadIdx.x;
  if (i < WS_FLOATS) ws[i] = 0.0f;
  if (i == 0) out[0] = 0.0f;
}

// ------ kernel 1: row norms, column sums, (optional) bf16 hi/lo planes ------
__global__ __launch_bounds__(256) void mmd_pre_kernel(const float* __restrict__ src,
                                                      const float* __restrict__ tgt,
                                                      float* __restrict__ ws,
                                                      unsigned short* __restrict__ HI,
                                                      unsigned short* __restrict__ LO) {
  __shared__ float red[256];
  const int i = blockIdx.x;      // row 0..8191
  const int t = threadIdx.x;     // dim 0..255
  float x = row_ptr(src, tgt, i)[t];
  if (HI) {
    unsigned short h = bf16_rne(x);
    HI[(size_t)i * DIMS + t] = h;
    LO[(size_t)i * DIMS + t] = bf16_rne(x - bf16_up(h));
  }
  atomicAdd(&ws[WS_COLSUM + t], x);
  red[t] = x * x;
  __syncthreads();
  for (int s = 128; s > 0; s >>= 1) {
    if (t < s) red[t] += red[t + s];
    __syncthreads();
  }
  if (t == 0) {
    ws[WS_SQ + i] = red[0];
    atomicAdd(&ws[WS_SUMSQ], red[0]);
  }
}

// ----------------- kernel 2: bandwidth -> exp coefficients ------------------
__global__ __launch_bounds__(256) void mmd_bw_kernel(float* __restrict__ ws) {
  __shared__ float red[256];
  const int t = threadIdx.x;
  float s = ws[WS_COLSUM + t];
  red[t] = s * s;
  __syncthreads();
  for (int k = 128; k > 0; k >>= 1) {
    if (t < k) red[t] += red[t + k];
    __syncthreads();
  }
  if (t == 0) {
    const float n = (float)TOT;
    float sumL2 = 2.0f * n * ws[WS_SUMSQ] - 2.0f * red[0];
    float bw = (sumL2 / (n * n - n)) / 4.0f;    // / KERNEL_MUL^(KERNEL_NUM//2)
    float b = bw;
    #pragma unroll
    for (int k = 0; k < 5; ++k) { ws[WS_COEF + k] = -1.0f / b; b *= 2.0f; }
  }
}

// -------- shared device code: one 32-K chunk of WMMA + the epilogue ---------
__device__ __forceinline__ void gram_chunk(const unsigned short* __restrict__ lds,
                                           int wave, int mn, int half, v8f acc[4]) {
  const unsigned short* Ah = lds;
  const unsigned short* Al = lds + SURF;
  const unsigned short* Bh = lds + 2 * SURF;
  const unsigned short* Bl = lds + 3 * SURF;

  // A operand (16x32 bf16, ISA 7.12.2): lane half h needs k in
  // [8h,8h+8) and [16+8h,16+8h+8) -> two contiguous 16B chunks.
  const int arow = wave * 16 + mn;
  v8u ahu, alu;
  {
    const unsigned short* a0 = Ah + arow * LSTRIDE + 8 * half;
    const unsigned short* a1 = Al + arow * LSTRIDE + 8 * half;
    ((v4u*)&ahu)[0] = *(const v4u*)(a0);
    ((v4u*)&ahu)[1] = *(const v4u*)(a0 + 16);
    ((v4u*)&alu)[0] = *(const v4u*)(a1);
    ((v4u*)&alu)[1] = *(const v4u*)(a1 + 16);
  }
  const v16bf ah = __builtin_bit_cast(v16bf, ahu);
  const v16bf al = __builtin_bit_cast(v16bf, alu);

  #pragma unroll
  for (int n = 0; n < 4; ++n) {
    // B operand (32x16 bf16): lane half h needs k in [8h,8h+8) and [16+8h,16+8h+8).
    const int bcol = n * 16 + mn;
    v8u bhu, blu;
    const unsigned short* b0 = Bh + bcol * LSTRIDE + 8 * half;
    const unsigned short* b1 = Bl + bcol * LSTRIDE + 8 * half;
    ((v4u*)&bhu)[0] = *(const v4u*)(b0);
    ((v4u*)&bhu)[1] = *(const v4u*)(b0 + 16);
    ((v4u*)&blu)[0] = *(const v4u*)(b1);
    ((v4u*)&blu)[1] = *(const v4u*)(b1 + 16);
    const v16bf bh = __builtin_bit_cast(v16bf, bhu);
    const v16bf bl = __builtin_bit_cast(v16bf, blu);
    // x.y ~= xh.yh + xh.yl + xl.yh
    acc[n] = __builtin_amdgcn_wmma_f32_16x16x32_bf16(false, ah, false, bh,
                                                     (short)0, acc[n], false, false);
    acc[n] = __builtin_amdgcn_wmma_f32_16x16x32_bf16(false, ah, false, bl,
                                                     (short)0, acc[n], false, false);
    acc[n] = __builtin_amdgcn_wmma_f32_16x16x32_bf16(false, al, false, bh,
                                                     (short)0, acc[n], false, false);
  }
}

__device__ __forceinline__ void gram_epilogue(const v8f acc[4], float* __restrict__ ws,
                                              int rowBase, int colBase,
                                              int wave, int mn, int half,
                                              int tid, float* __restrict__ red) {
  const float* sq = ws + WS_SQ;
  float c0 = ws[WS_COEF + 0], c1 = ws[WS_COEF + 1], c2 = ws[WS_COEF + 2];
  float c3 = ws[WS_COEF + 3], c4 = ws[WS_COEF + 4];

  const int grBase = rowBase + wave * 16 + half * 8;   // C layout: M = r + 8*half
  float sqr[8]; bool rsrc[8];
  #pragma unroll
  for (int r = 0; r < 8; ++r) {
    sqr[r]  = sq[grBase + r];
    rsrc[r] = (grBase + r) < N_ROWS;
  }

  float local = 0.0f;
  #pragma unroll
  for (int n = 0; n < 4; ++n) {
    const int gc = colBase + n * 16 + mn;
    const float sqc = sq[gc];
    const bool csrc = gc < N_ROWS;
    #pragma unroll
    for (int r = 0; r < 8; ++r) {
      float L2 = sqr[r] + sqc - 2.0f * acc[n][r];
      float v = __expf(c0 * L2) + __expf(c1 * L2) + __expf(c2 * L2) +
                __expf(c3 * L2) + __expf(c4 * L2);
      local += (rsrc[r] == csrc) ? v : -v;
    }
  }

  red[tid] = local;
  __syncthreads();
  for (int s = 64; s > 0; s >>= 1) {
    if (tid < s) red[tid] += red[tid + s];
    __syncthreads();
  }
  if (tid == 0) atomicAdd(&ws[WS_RES], red[0]);
}

// ---- kernel 3a: fast path — async double-buffered staging from HI/LO ------
__global__ __launch_bounds__(128) void mmd_gram_async_kernel(
    const unsigned short* __restrict__ HI, const unsigned short* __restrict__ LO,
    float* __restrict__ ws) {
  __shared__ __align__(16) unsigned short buf[2][CHUNK];
  __shared__ float red[128];

  const int tid  = threadIdx.x;
  const int wave = tid >> 5;
  const int lane = tid & 31;
  const int half = lane >> 4;
  const int mn   = lane & 15;

  const int ntile   = TOT / TILE;
  const int bx      = blockIdx.x % ntile;
  const int by      = blockIdx.x / ntile;
  const int rowBase = by * TILE;
  const int colBase = bx * TILE;

  // issue one chunk: 4 surfaces x 64 rows x 4 x 16B = 8 async ops per thread
  auto issue = [&](int kc, int b) {
    for (int i = tid; i < 4 * TILE * 4; i += 128) {
      const int surf = i >> 8;          // 0=Ahi 1=Alo 2=Bhi 3=Blo
      const int rem  = i & 255;
      const int row  = rem >> 2;
      const int seg  = rem & 3;         // 16B segment within 64B row-chunk
      const int grow = ((surf & 2) ? colBase : rowBase) + row;
      const unsigned short* g =
          ((surf & 1) ? LO : HI) + (size_t)grow * DIMS + kc + seg * 8;
      unsigned short* l = &buf[b][surf * SURF + row * LSTRIDE + seg * 8];
      async_copy16(g, l);
    }
  };

  v8f acc[4] = {};

  issue(0, 0);
  const int NCH = DIMS / KC;    // 8
  for (int c = 0; c < NCH; ++c) {
    const int cur = c & 1;
    if (c + 1 < NCH) { issue((c + 1) * KC, cur ^ 1); wait_async_le8(); }
    else             { wait_async_le0(); }
    __syncthreads();                        // chunk c visible to all waves
    gram_chunk(&buf[cur][0], wave, mn, half, acc);
    __syncthreads();                        // all waves done before buf reuse
  }

  gram_epilogue(acc, ws, rowBase, colBase, wave, mn, half, tid, red);
}

// ---- kernel 3b: fallback — convert fp32->bf16 hi/lo during staging ---------
__global__ __launch_bounds__(128) void mmd_gram_kernel(const float* __restrict__ src,
                                                       const float* __restrict__ tgt,
                                                       float* __restrict__ ws) {
  __shared__ __align__(16) unsigned short buf[CHUNK];
  __shared__ float red[128];

  const int tid  = threadIdx.x;
  const int wave = tid >> 5;
  const int lane = tid & 31;
  const int half = lane >> 4;
  const int mn   = lane & 15;

  const int ntile   = TOT / TILE;
  const int bx      = blockIdx.x % ntile;
  const int by      = blockIdx.x / ntile;
  const int rowBase = by * TILE;
  const int colBase = bx * TILE;

  v8f acc[4] = {};

  for (int kc = 0; kc < DIMS; kc += KC) {
    __syncthreads();
    for (int idx = tid; idx < TILE * KC; idx += 128) {
      const int r = idx >> 5, k = idx & 31;
      float xa = row_ptr(src, tgt, rowBase + r)[kc + k];
      unsigned short ha = bf16_rne(xa);
      buf[0 * SURF + r * LSTRIDE + k] = ha;
      buf[1 * SURF + r * LSTRIDE + k] = bf16_rne(xa - bf16_up(ha));
      float xb = row_ptr(src, tgt, colBase + r)[kc + k];
      unsigned short hb = bf16_rne(xb);
      buf[2 * SURF + r * LSTRIDE + k] = hb;
      buf[3 * SURF + r * LSTRIDE + k] = bf16_rne(xb - bf16_up(hb));
    }
    __syncthreads();
    gram_chunk(buf, wave, mn, half, acc);
  }

  gram_epilogue(acc, ws, rowBase, colBase, wave, mn, half, tid, red);
}

// ----------------------------- kernel 4: finalize ---------------------------
__global__ void mmd_fin_kernel(const float* __restrict__ ws, float* __restrict__ out) {
  if (threadIdx.x == 0 && blockIdx.x == 0) {
    const float inv_b2 = 1.0f / ((float)N_ROWS * (float)N_ROWS);
    out[0] = ws[WS_RES] * inv_b2;
  }
}

// ---------------------------------------------------------------------------
extern "C" void kernel_launch(void* const* d_in, const int* in_sizes, int n_in,
                              void* d_out, int out_size, void* d_ws, size_t ws_size,
                              hipStream_t stream) {
  (void)in_sizes; (void)n_in; (void)out_size;
  const float* src = (const float*)d_in[0];
  const float* tgt = (const float*)d_in[1];
  float* out = (float*)d_out;
  float* ws  = (float*)d_ws;

  const size_t base_bytes = (size_t)WS_FLOATS * sizeof(float);
  const bool fast = ws_size >= base_bytes + WS_HILO_BYTES;
  unsigned short* HI = fast ? (unsigned short*)((char*)d_ws + base_bytes) : nullptr;
  unsigned short* LO = fast ? HI + (size_t)TOT * DIMS : nullptr;

  const int zgrid = (WS_FLOATS + 255) / 256;
  mmd_zero_kernel<<<zgrid, 256, 0, stream>>>(ws, out);
  mmd_pre_kernel<<<TOT, 256, 0, stream>>>(src, tgt, ws, HI, LO);
  mmd_bw_kernel<<<1, 256, 0, stream>>>(ws);

  const int ntile = TOT / TILE;                  // 128
  if (fast) {
    mmd_gram_async_kernel<<<ntile * ntile, 128, 0, stream>>>(HI, LO, ws);
  } else {
    mmd_gram_kernel<<<ntile * ntile, 128, 0, stream>>>(src, tgt, ws);
  }
  mmd_fin_kernel<<<1, 1, 0, stream>>>(ws, out);
}
